// CoxPath_5231270166956
// MI455X (gfx1250) — matrix operand order repaired and checked by reading, not compiled, not run on hardware
//
#include <hip/hip_runtime.h>
#include <math.h>

// Problem constants (match reference)
#define Bdim 128
#define Pdim 2048
#define Fdim 512
#define Hdim 256
#define Cdim 16

// GEMM tiling: block tile 128x128, 8 waves of 32x64, K-step 32
#define MT 128
#define NTT 128
#define KT 32

#if defined(__has_builtin)
# if __has_builtin(__builtin_amdgcn_global_load_async_to_lds_b128)
#  define HAVE_ASYNC_LDS 1
# endif
#endif
#ifndef HAVE_ASYNC_LDS
# define HAVE_ASYNC_LDS 0
#endif

typedef __attribute__((ext_vector_type(16))) __bf16 v16bf;
typedef __attribute__((ext_vector_type(8)))  float  v8f;
typedef __attribute__((ext_vector_type(4)))  int    v4i;

// pointer-to-int4 in global (AS1) / LDS (AS3) address spaces for the async builtin
typedef __attribute__((address_space(1))) v4i* gptr_b128;
typedef __attribute__((address_space(3))) v4i* lptr_b128;

union FragAB { v16bf v; unsigned long long q[4]; };
union FragC  { v8f v; float f[8]; };

static __device__ __forceinline__ unsigned short f2bf(float f) {
    unsigned u = __float_as_uint(f);
    u += 0x7FFFu + ((u >> 16) & 1u);   // round-to-nearest-even
    return (unsigned short)(u >> 16);
}
static __device__ __forceinline__ float bf2f(unsigned short s) {
    return __uint_as_float(((unsigned)s) << 16);
}

static __device__ __forceinline__ void wait_async_all() {
#if HAVE_ASYNC_LDS
# if defined(__has_builtin) && __has_builtin(__builtin_amdgcn_s_wait_asynccnt)
    __builtin_amdgcn_s_wait_asynccnt(0);
# else
    asm volatile("s_wait_asynccnt 0x0" ::: "memory");
# endif
#endif
}

static __device__ __forceinline__ void async_b128(const unsigned short* g, unsigned short* l) {
#if HAVE_ASYNC_LDS
    __builtin_amdgcn_global_load_async_to_lds_b128(
        (gptr_b128)const_cast<unsigned short*>(g), (lptr_b128)l, 0, 0);
#endif
}

__global__ __launch_bounds__(256) void f32_to_bf16_kernel(
    const float* __restrict__ in, unsigned short* __restrict__ out, long long n4) {
    long long i = (long long)blockIdx.x * 256 + threadIdx.x;   // float4 quads
    if (i < n4) {
        float4 v = ((const float4*)in)[i];
        ushort4 o;
        o.x = f2bf(v.x); o.y = f2bf(v.y); o.z = f2bf(v.z); o.w = f2bf(v.w);
        ((ushort4*)out)[i] = o;
    }
}

// out[c][r] = bf16(in[r][c]) : produce transposed bf16 weights
__global__ __launch_bounds__(256) void f32_to_bf16_T_kernel(
    const float* __restrict__ in, unsigned short* __restrict__ out, int R, int Ccols) {
    long long idx = (long long)blockIdx.x * 256 + threadIdx.x;
    if (idx < (long long)R * Ccols) {
        int r = (int)(idx / Ccols);
        int c = (int)(idx % Ccols);
        out[(long long)c * R + r] = f2bf(in[idx]);
    }
}

// D = epilogue( A[M,K] @ (Bt[N,K])^T ).  A: bf16 (async-staged) or f32.
// Bt is B TRANSPOSED in memory, row stride ldb  -> async-staged directly as [n][k].
// D_TRANS=1: store D^T with row stride ldd (vectorized b128 stores).
// D_TRANS=0: store D row-major [M,N] (ldd = N).
// Batched via blockIdx.z with element strides sA/sB/sD.  BIAS_TANH: tanh(acc + bias[n]).
template<bool A_IS_F32, bool BIAS_TANH, bool D_TRANS>
__global__ __launch_bounds__(256) void gemm_bf16_kernel(
    const void* __restrict__ Av, const unsigned short* __restrict__ Bt,
    unsigned short* __restrict__ D, const float* __restrict__ bias,
    int K, int N,
    long long ldb, long long ldd,
    long long sA, long long sB, long long sD)
{
    __shared__ unsigned short lA[2][MT][KT];    // [m][k], k contiguous (16KB)
    __shared__ unsigned short lB[2][NTT][KT];   // [n][k], k contiguous (16KB)

    const int tid  = threadIdx.x;
    const int lane = tid & 31;
    const int w    = tid >> 5;              // 8 waves: 4 (M) x 2 (N)
    const int wm   = (w & 3) * 32;
    const int wn   = (w >> 2) * 64;
    const int m0   = blockIdx.x * MT;
    const int n0   = blockIdx.y * NTT;
    const long long zb = blockIdx.z;

    const unsigned short* Bp = Bt + zb * sB;
    unsigned short*       Dp = D  + zb * sD;

    FragC acc[2][4];
    #pragma unroll
    for (int i = 0; i < 2; ++i)
        #pragma unroll
        for (int j = 0; j < 4; ++j)
            #pragma unroll
            for (int r = 0; r < 8; ++r) acc[i][j].f[r] = 0.0f;

    const int fr  = lane & 15;              // fragment row (A: M) / col (B: N)
    const int kbA = (lane >> 4) * 8;        // A half-lane K base (interleaved layout)
    const int kbB = (lane >> 4) * 8;        // B half-lane K base (interleaved layout)

    // staging registers (f32-A path and non-async fallback)
    float4 aRegF[4];
    uint4  aReg[2];
    uint4  bReg[2];

    auto issue = [&](int k0, int c) {
        // ---- A tile 128x32 (row-major, lda == K) ----
        if constexpr (A_IS_F32) {
            const float* Ap = (const float*)Av + zb * sA;
            #pragma unroll
            for (int i = 0; i < 4; ++i) {
                int idx = tid + i * 256;          // 1024 float4 quads
                int r = idx >> 3;
                int cc = (idx & 7) * 4;
                aRegF[i] = *(const float4*)(Ap + (long long)(m0 + r) * K + k0 + cc);
            }
        } else {
            const unsigned short* Ap = (const unsigned short*)Av + zb * sA;
            #pragma unroll
            for (int i = 0; i < 2; ++i) {
                int idx = tid + i * 256;          // 512 b128 quads
                int r = idx >> 2;
                int cc = (idx & 3) * 8;
                const unsigned short* g = Ap + (long long)(m0 + r) * K + k0 + cc;
#if HAVE_ASYNC_LDS
                async_b128(g, &lA[c][r][cc]);
#else
                aReg[i] = *(const uint4*)g;
#endif
            }
        }
        // ---- B tile: rows n of Bt (already [n][k]) ----
        #pragma unroll
        for (int i = 0; i < 2; ++i) {
            int idx = tid + i * 256;              // 512 b128 quads
            int n = idx >> 2;                     // 0..127
            int cc = (idx & 3) * 8;               // 0..24
            const unsigned short* g = Bp + (long long)(n0 + n) * ldb + k0 + cc;
#if HAVE_ASYNC_LDS
            async_b128(g, &lB[c][n][cc]);
#else
            bReg[i] = *(const uint4*)g;
#endif
        }
        (void)c;
    };

    auto commit = [&](int c) {
        if constexpr (A_IS_F32) {
            #pragma unroll
            for (int i = 0; i < 4; ++i) {
                int idx = tid + i * 256;
                int r = idx >> 3;
                int cc = (idx & 7) * 4;
                unsigned short* dst = &lA[c][r][cc];
                dst[0] = f2bf(aRegF[i].x); dst[1] = f2bf(aRegF[i].y);
                dst[2] = f2bf(aRegF[i].z); dst[3] = f2bf(aRegF[i].w);
            }
        }
#if HAVE_ASYNC_LDS
        wait_async_all();                         // covers B tile (and A when bf16)
#else
        if constexpr (!A_IS_F32) {
            #pragma unroll
            for (int i = 0; i < 2; ++i) {
                int idx = tid + i * 256;
                int r = idx >> 2;
                int cc = (idx & 3) * 8;
                *(uint4*)&lA[c][r][cc] = aReg[i];
            }
        }
        #pragma unroll
        for (int i = 0; i < 2; ++i) {
            int idx = tid + i * 256;
            int n = idx >> 2;
            int cc = (idx & 3) * 8;
            *(uint4*)&lB[c][n][cc] = bReg[i];
        }
#endif
    };

    const int nk = K / KT;
    issue(0, 0);

    for (int kt = 0; kt < nk; ++kt) {
        const int c = kt & 1;
        commit(c);
        __syncthreads();
        if (kt + 1 < nk) issue((kt + 1) * KT, c ^ 1);

        // ---- fragments from LDS buffer c ----
        FragAB aF[2], bF[4];
        #pragma unroll
        for (int i = 0; i < 2; ++i) {
            const unsigned short* base = &lA[c][wm + i * 16 + fr][kbA];
            aF[i].q[0] = *(const unsigned long long*)(base + 0);   // K kb+0..3
            aF[i].q[1] = *(const unsigned long long*)(base + 4);   // K kb+4..7
            aF[i].q[2] = *(const unsigned long long*)(base + 16);  // K kb+16..19
            aF[i].q[3] = *(const unsigned long long*)(base + 20);  // K kb+20..23
        }
        #pragma unroll
        for (int j = 0; j < 4; ++j) {
            const unsigned short* base = &lB[c][wn + j * 16 + fr][kbB];
            bF[j].q[0] = *(const unsigned long long*)(base + 0);   // K kb+0..7, kb+16..23
            bF[j].q[1] = *(const unsigned long long*)(base + 4);
            bF[j].q[2] = *(const unsigned long long*)(base + 16);
            bF[j].q[3] = *(const unsigned long long*)(base + 20);
        }
        #pragma unroll
        for (int i = 0; i < 2; ++i)
            #pragma unroll
            for (int j = 0; j < 4; ++j)
                acc[i][j].v = __builtin_amdgcn_wmma_f32_16x16x32_bf16(
                    false, aF[i].v, false, bF[j].v,
                    (short)0, acc[i][j].v, false, false);
    }

    // ---- epilogue ----
    const int col     = lane & 15;
    const int rowHalf = (lane >> 4) * 8;
    #pragma unroll
    for (int i = 0; i < 2; ++i) {
        #pragma unroll
        for (int j = 0; j < 4; ++j) {
            int gn = n0 + wn + j * 16 + col;
            float bv = BIAS_TANH ? bias[gn] : 0.0f;
            if constexpr (D_TRANS) {
                // v8f = 8 consecutive M rows at fixed N -> one b128 store into D^T
                int gm0 = m0 + wm + i * 16 + rowHalf;
                uint4 o;
                unsigned short* os = (unsigned short*)&o;
                #pragma unroll
                for (int r = 0; r < 8; ++r) {
                    float v = acc[i][j].f[r];
                    if (BIAS_TANH) v = tanhf(v + bv);
                    os[r] = f2bf(v);
                }
                *(uint4*)(Dp + (long long)gn * ldd + gm0) = o;
            } else {
                #pragma unroll
                for (int r = 0; r < 8; ++r) {
                    int gm = m0 + wm + i * 16 + rowHalf + r;
                    float v = acc[i][j].f[r];
                    if (BIAS_TANH) v = tanhf(v + bv);
                    Dp[(long long)gm * N + gn] = f2bf(v);
                }
            }
        }
    }
}

// h2T layout: [H][B*P].  out[b] = sum_p tanh(sum_h h2T[h][b*P+p]*lw1[h] + lb1)*lw2[p]
//                                + sum_c clinical[b,c]*lw2[P+c] + lb2
__global__ __launch_bounds__(256) void finalize_kernel(
    const unsigned short* __restrict__ h2T, const float* __restrict__ clinical,
    const float* __restrict__ lw1, const float* __restrict__ lb1,
    const float* __restrict__ lw2, const float* __restrict__ lb2,
    float* __restrict__ out)
{
    __shared__ float slw1[Hdim];
    __shared__ float red[256];
    const int b = blockIdx.x;
    const int tid = threadIdx.x;
    const long long BP = (long long)Bdim * Pdim;
    slw1[tid] = lw1[tid];
    __syncthreads();

    const float lb1v = lb1[0];
    float local = 0.0f;
    for (int p = tid; p < Pdim; p += 256) {
        const unsigned short* colp = h2T + (long long)b * Pdim + p;
        float acc = 0.0f;
        #pragma unroll 8
        for (int h = 0; h < Hdim; ++h)
            acc += bf2f(colp[(long long)h * BP]) * slw1[h];   // coalesced across tid
        local += tanhf(acc + lb1v) * lw2[p];
    }
    red[tid] = local;
    __syncthreads();
    for (int s = 128; s > 0; s >>= 1) {
        if (tid < s) red[tid] += red[tid + s];
        __syncthreads();
    }
    if (tid == 0) {
        float tot = red[0];
        #pragma unroll
        for (int c = 0; c < Cdim; ++c)
            tot += clinical[b * Cdim + c] * lw2[Pdim + c];
        out[b] = tot + lb2[0];
    }
}

extern "C" void kernel_launch(void* const* d_in, const int* in_sizes, int n_in,
                              void* d_out, int out_size, void* d_ws, size_t ws_size,
                              hipStream_t stream) {
    const float* x        = (const float*)d_in[0];
    const float* adj      = (const float*)d_in[1];
    const float* clinical = (const float*)d_in[2];
    const float* W1       = (const float*)d_in[3];
    const float* b1       = (const float*)d_in[4];
    const float* W2       = (const float*)d_in[5];
    const float* b2       = (const float*)d_in[6];
    const float* lw1      = (const float*)d_in[7];
    const float* lb1      = (const float*)d_in[8];
    const float* lw2      = (const float*)d_in[9];
    const float* lb2      = (const float*)d_in[10];
    float* out            = (float*)d_out;

    const long long BP   = (long long)Bdim * Pdim;      // 262144
    const long long nAdj = (long long)Pdim * Pdim;      // 4M
    const long long nW1  = (long long)Fdim * Hdim;
    const long long nW2  = (long long)Hdim * Hdim;
    const long long nAct = BP * Hdim;                   // 64M elems

    char* ws = (char*)d_ws;
    unsigned short* adjb = (unsigned short*)ws;  ws += ((nAdj * 2 + 255) / 256) * 256;
    unsigned short* W1T  = (unsigned short*)ws;  ws += ((nW1  * 2 + 255) / 256) * 256;  // [H][F]
    unsigned short* W2T  = (unsigned short*)ws;  ws += ((nW2  * 2 + 255) / 256) * 256;  // [H][H]
    unsigned short* supT = (unsigned short*)ws;  ws += ((nAct * 2 + 255) / 256) * 256;  // [H][B*P]
    unsigned short* hbuf = (unsigned short*)ws;  // h1 [B*P][H], then h2T [H][B*P]

    // 1) operand preparation
    f32_to_bf16_kernel<<<(unsigned)((nAdj / 4 + 255) / 256), 256, 0, stream>>>(adj, adjb, nAdj / 4);
    f32_to_bf16_T_kernel<<<(unsigned)((nW1 + 255) / 256), 256, 0, stream>>>(W1, W1T, Fdim, Hdim);
    f32_to_bf16_T_kernel<<<(unsigned)((nW2 + 255) / 256), 256, 0, stream>>>(W2, W2T, Hdim, Hdim);

    // 2) sup1T = (x @ W1)^T          A=x f32 [B*P,F], Bt=W1T [H][F], D^T [H][B*P]
    {
        dim3 grid((unsigned)(BP / MT), Hdim / NTT, 1);
        gemm_bf16_kernel<true, false, true><<<grid, 256, 0, stream>>>(
            (const void*)x, W1T, supT, nullptr,
            Fdim, Hdim, (long long)Fdim, BP, 0LL, 0LL, 0LL);
    }
    // 3) h1 = tanh(adj @ sup1 + b1)  A=adjb [P,P], Bt=sup1T (+z*P, ldb=BP), D normal [P,H]
    {
        dim3 grid(Pdim / MT, Hdim / NTT, Bdim);
        gemm_bf16_kernel<false, true, false><<<grid, 256, 0, stream>>>(
            (const void*)adjb, supT, hbuf, b1,
            Pdim, Hdim, BP, (long long)Hdim,
            0LL, (long long)Pdim, (long long)Pdim * Hdim);
    }
    // 4) sup2T = (h1 @ W2)^T         A=h1 [B*P,H], Bt=W2T [H][H], D^T [H][B*P]
    {
        dim3 grid((unsigned)(BP / MT), Hdim / NTT, 1);
        gemm_bf16_kernel<false, false, true><<<grid, 256, 0, stream>>>(
            (const void*)hbuf, W2T, supT, nullptr,
            Hdim, Hdim, (long long)Hdim, BP, 0LL, 0LL, 0LL);
    }
    // 5) h2T = tanh(adj @ sup2 + b2)^T   A=adjb, Bt=sup2T (+z*P), D^T [H][B*P] (+z*P)
    {
        dim3 grid(Pdim / MT, Hdim / NTT, Bdim);
        gemm_bf16_kernel<false, true, true><<<grid, 256, 0, stream>>>(
            (const void*)adjb, supT, hbuf, b2,
            Pdim, Hdim, BP, BP,
            0LL, (long long)Pdim, (long long)Pdim);
    }
    // 6) head: lin1 + tanh + concat(clinical) + lin2
    finalize_kernel<<<Bdim, 256, 0, stream>>>(hbuf, clinical, lw1, lb1, lw2, lb2, out);
}
